// ViTBlock_54786602828543
// MI455X (gfx1250) — matrix operand order repaired and checked by reading, not compiled, not run on hardware
//
#include <hip/hip_runtime.h>

// ---------------------------------------------------------------------------
// Fused windowed-attention ViT block for MI455X (gfx1250, wave32, WMMA).
// One workgroup per 16x16 window (1024 blocks), 128 threads = 4 waves.
// All GEMMs via v_wmma_f32_16x16x32_bf16 (fp32 accumulate).
// LDS (exactly 320KB): X bf16 96K | K bf16 96K | V^T bf16 96K | 4x8K stage.
// ---------------------------------------------------------------------------

typedef __attribute__((ext_vector_type(16))) __bf16 v16bf;
typedef __attribute__((ext_vector_type(8)))  float  v8f;

union Frag {
  v16bf        v;
  unsigned int u[8];
  __bf16       h[16];
};

union Pack4 {                 // 8 bf16 <-> one 16-byte store
  uint4  u4;
  __bf16 h[8];
};

__device__ __forceinline__ v8f wmma_bf16(const Frag& a, const Frag& b, v8f c) {
  // 8 args: (neg_a, A, neg_b, B, c_mod, C, reuse_a, reuse_b)
  return __builtin_amdgcn_wmma_f32_16x16x32_bf16(false, a.v, false, b.v,
                                                 (short)0, c, false, false);
}

__device__ __forceinline__ v8f vzero8() {
  v8f z = {0.f, 0.f, 0.f, 0.f, 0.f, 0.f, 0.f, 0.f};
  return z;
}

// A fragment (16x32, bf16) from row-major LDS matrix, row stride ld elements.
// Layout: lanes 0-15: M=lane, K=0..15; lanes 16-31: K=16..31 (matches B).
__device__ __forceinline__ Frag load_a_lds(const __bf16* base, int r0, int ld,
                                           int k0, int ln, int half) {
  Frag a;
  const __bf16* row = base + (r0 + ln) * ld + k0;
#pragma unroll
  for (int j = 0; j < 8; ++j)
    a.u[j] = *(const unsigned int*)(row + 16 * half + 2 * j);
  return a;
}

// B fragment (32x16, bf16) from LDS where element (k,n) sits at base[n*ld + k]
// (K pairs contiguous). Contiguous K order (as A): lane n=ln, VGPR j packs K={16*half+2j, +1}.
__device__ __forceinline__ Frag load_b_lds(const __bf16* base, int n0, int ld,
                                           int k0, int ln, int half) {
  Frag b;
  const __bf16* col = base + (n0 + ln) * ld + k0 + 16 * half;
#pragma unroll
  for (int j = 0; j < 8; ++j)
    b.u[j] = *(const unsigned int*)(col + 2 * j);
  return b;
}

// B fragment from a row-major f32 weight matrix W[n][k] (ldk=192), converting
// f32 -> bf16 on the fly (weights are tiny and L2-resident; pairs -> b64).
__device__ __forceinline__ Frag load_b_gf32(const float* __restrict__ w, int n0,
                                            int k0, int ln, int half) {
  Frag b;
  const float* row = w + (n0 + ln) * 192;
#pragma unroll
  for (int j = 0; j < 8; ++j) {
    int k = k0 + 16 * half + 2 * j;
    b.h[2 * j]     = (__bf16)row[k];
    b.h[2 * j + 1] = (__bf16)row[k + 1];
  }
  return b;
}

#define LDS_WAIT() asm volatile("s_wait_dscnt 0x0" ::: "memory")

__global__ void __launch_bounds__(128, 1)
vit_window_attn_kernel(const float* __restrict__ x,
                       const float* __restrict__ w_qkv,
                       const float* __restrict__ b_qkv,
                       const float* __restrict__ w_pe,
                       const float* __restrict__ b_pe,
                       const float* __restrict__ w_out,
                       const float* __restrict__ b_out,
                       float* __restrict__ out) {
  extern __shared__ __align__(16) unsigned char smem[];
  __bf16* xs = (__bf16*)(smem);            // [256][192] tokens x channels
  __bf16* ks = (__bf16*)(smem + 98304);    // [256][192] keys x channels
  __bf16* vt = (__bf16*)(smem + 196608);   // [192][256] channels x keys (V^T)
  const int tid  = threadIdx.x;
  const int wave = tid >> 5;
  const int lane = tid & 31;
  const int ln   = lane & 15;
  const int half = lane >> 4;
  __bf16* st = (__bf16*)(smem + 294912) + wave * 4096;  // 8KB per-wave stage

  const int nw = blockIdx.x;               // window id 0..1023
  const int bN = nw >> 8;                  // batch
  const int hw = (nw >> 4) & 15;           // window row
  const int ww = nw & 15;                  // window col

  // ---- Phase 0: gather window of x (NCHW) -> LDS bf16 [t][c], b128 loads --
  for (int e = tid; e < 12288; e += 128) {
    int idx = e << 2;                      // element index, 4 per thread
    int t0  = idx & 255;                   // token (multiple of 4)
    int c   = idx >> 8;                    // channel
    int i = t0 >> 4, j = t0 & 15;          // j in {0,4,8,12}: 16B aligned
    const float4 xv =
        *(const float4*)(x + (((bN * 192 + c) * 256) + hw * 16 + i) * 256 +
                         ww * 16 + j);
    __bf16* dst = xs + t0 * 192 + c;
    dst[0]       = (__bf16)xv.x;
    dst[192]     = (__bf16)xv.y;
    dst[2 * 192] = (__bf16)xv.z;
    dst[3 * 192] = (__bf16)xv.w;
  }
  __syncthreads();

  // ---- Phase 1a: K = X Wk^T + bk -> ks[key][ch] (uniform store target) ----
  for (int qk = 0; qk < 3; ++qk) {
    int ntl = wave + 4 * qk;               // 0..11 over 4 waves
    int co0 = 192 + ntl * 16;              // w_qkv row base (K part)
    Frag B[6];
#pragma unroll
    for (int kt = 0; kt < 6; ++kt) B[kt] = load_b_gf32(w_qkv, co0, kt * 32, ln, half);
    float bq = b_qkv[co0 + ln];
    for (int mt = 0; mt < 16; ++mt) {
      Frag A[6];
#pragma unroll
      for (int kt = 0; kt < 6; ++kt)
        A[kt] = load_a_lds(xs, mt * 16, 192, kt * 32, ln, half);
      v8f acc = vzero8();
#pragma unroll
      for (int kt = 0; kt < 6; ++kt) acc = wmma_bf16(A[kt], B[kt], acc);
#pragma unroll
      for (int r = 0; r < 8; ++r) {
        int m = mt * 16 + r + 8 * half;    // C layout: rows r / r+8 per half
        ks[m * 192 + ntl * 16 + ln] = (__bf16)(acc[r] + bq);
      }
    }
  }
  // ---- Phase 1b: V = X Wv^T + bv -> vt[ch][key] (packed b128 stores) ------
  for (int qv = 0; qv < 3; ++qv) {
    int ntl = wave + 4 * qv;
    int co0 = 384 + ntl * 16;              // w_qkv row base (V part)
    Frag B[6];
#pragma unroll
    for (int kt = 0; kt < 6; ++kt) B[kt] = load_b_gf32(w_qkv, co0, kt * 32, ln, half);
    float bq = b_qkv[co0 + ln];
    for (int mt = 0; mt < 16; ++mt) {
      Frag A[6];
#pragma unroll
      for (int kt = 0; kt < 6; ++kt)
        A[kt] = load_a_lds(xs, mt * 16, 192, kt * 32, ln, half);
      v8f acc = vzero8();
#pragma unroll
      for (int kt = 0; kt < 6; ++kt) acc = wmma_bf16(A[kt], B[kt], acc);
      Pack4 pk;                            // rows m contiguous in vt layout
#pragma unroll
      for (int r = 0; r < 8; ++r) pk.h[r] = (__bf16)(acc[r] + bq);
      *(uint4*)(vt + (ntl * 16 + ln) * 256 + mt * 16 + 8 * half) = pk.u4;
    }
  }
  __syncthreads();

  const float scale = 0.07216878364870322f;  // 1/sqrt(192), folded into Q

  // ---- Phase 2: per wave, 4 blocks of 16 query rows -----------------------
  for (int rb = 0; rb < 4; ++rb) {
    int m0 = (wave * 4 + rb) * 16;

    // 2a: Q tile (16x192), bias + softmax scale fused, staged bf16
    {
      Frag Aq[6];
#pragma unroll
      for (int kt = 0; kt < 6; ++kt)
        Aq[kt] = load_a_lds(xs, m0, 192, kt * 32, ln, half);
      for (int nt = 0; nt < 12; ++nt) {
        Frag B[6];
#pragma unroll
        for (int kt = 0; kt < 6; ++kt)
          B[kt] = load_b_gf32(w_qkv, nt * 16, kt * 32, ln, half);
        v8f acc = vzero8();
#pragma unroll
        for (int kt = 0; kt < 6; ++kt) acc = wmma_bf16(Aq[kt], B[kt], acc);
        float bq = b_qkv[nt * 16 + ln];
#pragma unroll
        for (int r = 0; r < 8; ++r) {
          int ml = r + 8 * half;
          st[ml * 192 + nt * 16 + ln] = (__bf16)((acc[r] + bq) * scale);
        }
      }
    }
    LDS_WAIT();

    // 2b: S = Q K^T (16x256), exact softmax, P staged bf16 (16x256)
    {
      Frag Aq[6];
#pragma unroll
      for (int kt = 0; kt < 6; ++kt)
        Aq[kt] = load_a_lds(st, 0, 192, kt * 32, ln, half);
      v8f sf[16];
#pragma unroll
      for (int f = 0; f < 16; ++f) sf[f] = vzero8();
#pragma unroll
      for (int f = 0; f < 16; ++f) {
        Frag B[6];
#pragma unroll
        for (int kt = 0; kt < 6; ++kt)
          B[kt] = load_b_lds(ks, f * 16, 192, kt * 32, ln, half);
#pragma unroll
        for (int kt = 0; kt < 6; ++kt) sf[f] = wmma_bf16(Aq[kt], B[kt], sf[f]);
      }
      float mx[8], sm[8];
#pragma unroll
      for (int r = 0; r < 8; ++r) {
        float m = sf[0][r];
#pragma unroll
        for (int f = 1; f < 16; ++f) m = fmaxf(m, sf[f][r]);
#pragma unroll
        for (int off = 1; off < 16; off <<= 1)
          m = fmaxf(m, __shfl_xor(m, off, 32));
        mx[r] = m;
      }
#pragma unroll
      for (int r = 0; r < 8; ++r) {
        float s = 0.f;
#pragma unroll
        for (int f = 0; f < 16; ++f) {
          float e = __expf(sf[f][r] - mx[r]);
          sf[f][r] = e;
          s += e;
        }
#pragma unroll
        for (int off = 1; off < 16; off <<= 1) s += __shfl_xor(s, off, 32);
        sm[r] = 1.0f / s;
      }
#pragma unroll
      for (int f = 0; f < 16; ++f) {
#pragma unroll
        for (int r = 0; r < 8; ++r) {
          int ml = r + 8 * half;
          st[ml * 256 + f * 16 + ln] = (__bf16)(sf[f][r] * sm[r]);
        }
      }
    }
    LDS_WAIT();

    // 2c: O = P V (16x192)
    v8f oa[12];
#pragma unroll
    for (int nt = 0; nt < 12; ++nt) oa[nt] = vzero8();
    {
      Frag Ap[8];
#pragma unroll
      for (int kt = 0; kt < 8; ++kt)
        Ap[kt] = load_a_lds(st, 0, 256, kt * 32, ln, half);
#pragma unroll
      for (int nt = 0; nt < 12; ++nt) {
        Frag B[8];
#pragma unroll
        for (int kt = 0; kt < 8; ++kt)
          B[kt] = load_b_lds(vt, nt * 16, 256, kt * 32, ln, half);
#pragma unroll
        for (int kt = 0; kt < 8; ++kt) oa[nt] = wmma_bf16(Ap[kt], B[kt], oa[nt]);
      }
    }

    // 2d: + LePE (branchless depthwise 3x3 on raw-reshaped V image from vt),
    //     then restage (O+lepe) bf16 for the output projection
#pragma unroll
    for (int nt = 0; nt < 12; ++nt) {
#pragma unroll
      for (int r = 0; r < 8; ++r) {
        int ml = r + 8 * half;
        int t  = m0 + ml;
        int c  = nt * 16 + ln;
        int fi = t * 192 + c;              // flat index in (256,192) view
        int cc = fi >> 8;                  // channel of the (C,16,16) image
        int rm = fi & 255;
        int ai = rm >> 4, bi = rm & 15;
        float lp = b_pe[cc];
#pragma unroll
        for (int da = 0; da < 3; ++da) {
          int a2 = ai + da - 1;
          bool va = ((unsigned)a2 < 16u);
          int a2c = va ? a2 : ai;          // safe clamp, masked below
#pragma unroll
          for (int db = 0; db < 3; ++db) {
            int b2 = bi + db - 1;
            bool vb = va && ((unsigned)b2 < 16u);
            int b2c = ((unsigned)b2 < 16u) ? b2 : bi;
            int g  = cc * 256 + a2c * 16 + b2c;  // back into (256,192) flat
            int t2 = g / 192;
            int c2 = g - t2 * 192;
            float vv = (float)vt[c2 * 256 + t2];
            float wv = vb ? w_pe[cc * 9 + da * 3 + db] : 0.0f;
            lp = fmaf(wv, vv, lp);         // cndmask + fma, no branches
          }
        }
        st[ml * 192 + c] = (__bf16)(oa[nt][r] + lp);
      }
    }
    LDS_WAIT();

    // 2e: out projection (O+lepe) Wout^T + bout, reverse window partition,
    //     packed float4 output stores (8 consecutive j per lane)
    {
      Frag Ao[6];
#pragma unroll
      for (int kt = 0; kt < 6; ++kt)
        Ao[kt] = load_a_lds(st, 0, 192, kt * 32, ln, half);
      const int i0 = m0 >> 4;              // m0 is a multiple of 16
      float* obase = out + (((bN * 192) * 256) + hw * 16 + i0) * 256 +
                     ww * 16 + 8 * half;
#pragma unroll
      for (int nt = 0; nt < 12; ++nt) {
        Frag B[6];
#pragma unroll
        for (int kt = 0; kt < 6; ++kt)
          B[kt] = load_b_gf32(w_out, nt * 16, kt * 32, ln, half);
        v8f acc = vzero8();
#pragma unroll
        for (int kt = 0; kt < 6; ++kt) acc = wmma_bf16(Ao[kt], B[kt], acc);
        int co = nt * 16 + ln;
        float bo = b_out[co];
        float4 v0 = make_float4(acc[0] + bo, acc[1] + bo, acc[2] + bo, acc[3] + bo);
        float4 v1 = make_float4(acc[4] + bo, acc[5] + bo, acc[6] + bo, acc[7] + bo);
        float* p = obase + co * 65536;     // channel plane stride 256*256
        *(float4*)(p)     = v0;            // j = 8*half + 0..3
        *(float4*)(p + 4) = v1;            // j = 8*half + 4..7
      }
    }
  }
}

extern "C" void kernel_launch(void* const* d_in, const int* in_sizes, int n_in,
                              void* d_out, int out_size, void* d_ws, size_t ws_size,
                              hipStream_t stream) {
  (void)in_sizes; (void)n_in; (void)out_size; (void)d_ws; (void)ws_size;
  const float* x     = (const float*)d_in[0];
  const float* w_qkv = (const float*)d_in[1];
  const float* b_qkv = (const float*)d_in[2];
  const float* w_pe  = (const float*)d_in[3];
  const float* b_pe  = (const float*)d_in[4];
  const float* w_out = (const float*)d_in[5];
  const float* b_out = (const float*)d_in[6];
  float* out = (float*)d_out;

  const size_t shmem = 327680;  // 320 KB: full WGP LDS
  (void)hipFuncSetAttribute(reinterpret_cast<const void*>(vit_window_attn_kernel),
                            hipFuncAttributeMaxDynamicSharedMemorySize,
                            (int)shmem);
  vit_window_attn_kernel<<<dim3(1024), dim3(128), shmem, stream>>>(
      x, w_qkv, b_qkv, w_pe, b_pe, w_out, b_out, out);
}
